// Demucs_46548855554546
// MI455X (gfx1250) — matrix operand order repaired and checked by reading, not compiled, not run on hardware
//
#include <hip/hip_runtime.h>

// ---------------------------------------------------------------------------
// Types for CDNA5 WMMA / TDM
// ---------------------------------------------------------------------------
typedef __attribute__((ext_vector_type(16))) _Float16 v16h;
typedef __attribute__((ext_vector_type(8)))  _Float16 v8h;
typedef __attribute__((ext_vector_type(8)))  float    v8f;
typedef __attribute__((ext_vector_type(4)))  unsigned int u32x4;
typedef __attribute__((ext_vector_type(8)))  unsigned int u32x8;

#define BM 64
#define BN 64
#define BKT 32
#define BKP 40   // padded LDS row stride (halves): 80B, 16B aligned

// ---------------------------------------------------------------------------
// WMMA GEMM:  D[M,N](f16) = A(f16) x B(f16) (+bias[m] f32) (+relu)
//   TA=0: A is (M,K) row-major  -> A tile staged to LDS by the TENSOR DMA
//   TA=1: A is (K,M) row-major (accessed transposed)
//   TB=0: B is (K,N) row-major
//   TB=1: B is (N,K) row-major (accessed transposed)
//   EPI : 0 = none, 1 = +bias, 2 = +bias then relu
//
// 128 threads = 4 waves; each wave computes a 32x32 C tile -> 4 WMMAs per
// K-step reusing 2 A-frags x 2 B-frags (2 ds_load_b128 per WMMA).
// LDS double buffered: TDM/global loads of tile t+1 overlap WMMAs of tile t.
//
// INVARIANTS (host enforces via buffer padding):
//   M % 64 == 0, N % 64 == 0, K % 32 == 0
//   operand bases 16B aligned; lda/ldb multiples of 8 halves
// ---------------------------------------------------------------------------
template<int TA, int TB, int EPI>
__global__ __launch_bounds__(128)
void wmma_gemm_kernel(const _Float16* __restrict__ A, const _Float16* __restrict__ B,
                      _Float16* __restrict__ D, const float* __restrict__ bias,
                      int M, int N, int K, int lda, int ldb, int ldd,
                      long sA, long sB, long sD)
{
    __shared__ _Float16 As[2 * BM * BKP];
    __shared__ _Float16 Bs[2 * BN * BKP];

    A += (long)blockIdx.z * sA;
    B += (long)blockIdx.z * sB;
    D += (long)blockIdx.z * sD;

    const int tid   = threadIdx.x;
    const int lane  = tid & 31;
    const int wave  = tid >> 5;   // 0..3
    const int waveM = wave & 1;   // 32-row slice of C
    const int waveN = wave >> 1;  // 32-col slice of C
    const int row0  = blockIdx.y * BM;
    const int col0  = blockIdx.x * BN;

    v8f acc00 = {}, acc01 = {}, acc10 = {}, acc11 = {};

    auto stage = [&](int kk, int p) {
        _Float16* AsP = &As[p * BM * BKP];
        _Float16* BsP = &Bs[p * BN * BKP];
        // ---- A tile -> AsP[m][k] -----------------------------------------
        if (TA == 0) {
            // Tensor Data Mover: 64 x 32 f16 tile, row stride lda; LDS pad of
            // 4 dwords per 16 dwords reproduces the BKP=40 layout.
            if (wave == 0) {
                unsigned long long ga =
                    (unsigned long long)(const void*)(A + (long)row0 * lda + kk);
                unsigned ldsa = (unsigned)(unsigned long long)(const void*)AsP;
                unsigned td0 = (unsigned)(K - kk);
                unsigned td1 = (unsigned)(M - row0);
                u32x4 g0;
                g0[0] = 1u;                                               // count=1
                g0[1] = ldsa;                                             // lds_addr
                g0[2] = (unsigned)ga;                                     // ga[31:0]
                g0[3] = (unsigned)((ga >> 32) & 0x1FFFFFFu) | (2u << 30); // ga[56:32], type=2
                u32x8 g1;
                g1[0] = (1u << 16)   // data_size = 2B
                      | (1u << 20)   // pad_enable
                      | (3u << 22)   // pad_interval: 16 dwords
                      | (3u << 25);  // pad_amount : 4 dwords
                g1[1] = (td0 & 0xFFFFu) << 16;
                g1[2] = (td0 >> 16) | ((td1 & 0xFFFFu) << 16);
                g1[3] = (td1 >> 16) | ((unsigned)BKT << 16);              // tile_dim0=32
                g1[4] = (unsigned)BM;                                     // tile_dim1=64
                g1[5] = (unsigned)lda;                                    // dim0_stride
                g1[6] = 0u;
                g1[7] = 0u;
                asm volatile("tensor_load_to_lds %0, %1" :: "s"(g0), "s"(g1) : "memory");
            }
        } else {
            const int k  = tid >> 2;           // 0..31
            const int ms = (tid & 3) * 16;     // 0,16,32,48
            const _Float16* src = A + (long)(kk + k) * lda + (row0 + ms);
            v8h v0 = *(const v8h*)src;
            v8h v1 = *(const v8h*)(src + 8);
            #pragma unroll
            for (int i = 0; i < 8; ++i) {
                AsP[(ms + i) * BKP + k]     = v0[i];
                AsP[(ms + 8 + i) * BKP + k] = v1[i];
            }
        }
        // ---- B tile -> BsP[n][k] -----------------------------------------
        if (TB == 0) {
            const int k  = tid >> 2;
            const int ns = (tid & 3) * 16;
            const _Float16* src = B + (long)(kk + k) * ldb + (col0 + ns);
            v8h v0 = *(const v8h*)src;
            v8h v1 = *(const v8h*)(src + 8);
            #pragma unroll
            for (int i = 0; i < 8; ++i) {
                BsP[(ns + i) * BKP + k]     = v0[i];
                BsP[(ns + 8 + i) * BKP + k] = v1[i];
            }
        } else {
            const int n  = tid >> 1;           // 0..63
            const int ks = (tid & 1) * 16;     // 0,16
            const _Float16* src = B + (long)(col0 + n) * ldb + (kk + ks);
            v8h v0 = *(const v8h*)src;
            v8h v1 = *(const v8h*)(src + 8);
            *(v8h*)&BsP[n * BKP + ks]     = v0;   // ds_store_b128
            *(v8h*)&BsP[n * BKP + ks + 8] = v1;   // ds_store_b128
        }
        // speculative prefetch of the following B tile (global_prefetch_b8)
        if (kk + BKT < K) {
            const _Float16* pf = (TB == 0)
                ? (B + (long)(kk + BKT) * ldb + col0)
                : (B + (long)col0 * ldb + (kk + BKT));
            __builtin_prefetch(pf, 0, 0);
        }
    };

    const int nt = K / BKT;
    stage(0, 0);
    for (int t = 0; t < nt; ++t) {
        // TENSORcnt is per-wave: free no-op for the waves that issued nothing.
        if (TA == 0) __builtin_amdgcn_s_wait_tensorcnt(0);
        __syncthreads();
        if (t + 1 < nt) stage((t + 1) * BKT, (t + 1) & 1);

        const _Float16* AsP = &As[(t & 1) * BM * BKP];
        const _Float16* BsP = &Bs[(t & 1) * BN * BKP];

        // A 16x32 frags (ISA 7.12.2): lane m = lane&15, half = lane>>4,
        // elements 0..7 -> k = half*8+e ; 8..15 -> k = 16+half*8+e
        const int half  = lane >> 4;
        const int mrow0 = waveM * 32 + (lane & 15);
        const int mrow1 = mrow0 + 16;
        v16h afrag0, afrag1;
        {
            const v8h x0 = *(const v8h*)&AsP[mrow0 * BKP + half * 8];
            const v8h x1 = *(const v8h*)&AsP[mrow0 * BKP + 16 + half * 8];
            const v8h y0 = *(const v8h*)&AsP[mrow1 * BKP + half * 8];
            const v8h y1 = *(const v8h*)&AsP[mrow1 * BKP + 16 + half * 8];
            #pragma unroll
            for (int i = 0; i < 8; ++i) {
                afrag0[i] = x0[i]; afrag0[i + 8] = x1[i];
                afrag1[i] = y0[i]; afrag1[i + 8] = y1[i];
            }
        }
        // B 32x16 frags: lane n = lane&15, elements 0..7 -> k = half*8+e ; 8..15 -> k = 16+half*8+e
        const int ncol0 = waveN * 32 + (lane & 15);
        const int ncol1 = ncol0 + 16;
        const int kh    = half * 8;
        v16h bfrag0, bfrag1;
        {
            const v8h x0 = *(const v8h*)&BsP[ncol0 * BKP + kh];
            const v8h x1 = *(const v8h*)&BsP[ncol0 * BKP + kh + 16];
            const v8h y0 = *(const v8h*)&BsP[ncol1 * BKP + kh];
            const v8h y1 = *(const v8h*)&BsP[ncol1 * BKP + kh + 16];
            #pragma unroll
            for (int i = 0; i < 8; ++i) {
                bfrag0[i] = x0[i]; bfrag0[i + 8] = x1[i];
                bfrag1[i] = y0[i]; bfrag1[i + 8] = y1[i];
            }
        }

        acc00 = __builtin_amdgcn_wmma_f32_16x16x32_f16(
                    false, afrag0, false, bfrag0, (short)0, acc00, false, false);
        acc01 = __builtin_amdgcn_wmma_f32_16x16x32_f16(
                    false, afrag0, false, bfrag1, (short)0, acc01, false, false);
        acc10 = __builtin_amdgcn_wmma_f32_16x16x32_f16(
                    false, afrag1, false, bfrag0, (short)0, acc10, false, false);
        acc11 = __builtin_amdgcn_wmma_f32_16x16x32_f16(
                    false, afrag1, false, bfrag1, (short)0, acc11, false, false);
    }

    // ---- epilogue: unconditional (M,N multiples of 64) --------------------
    const int mb0 = row0 + waveM * 32 + (lane >> 4) * 8;
    const int nb0 = col0 + waveN * 32 + (lane & 15);
    #pragma unroll
    for (int r = 0; r < 8; ++r) {
        const int m0 = mb0 + r;
        const int m1 = m0 + 16;
        float b0 = (EPI >= 1) ? bias[m0] : 0.f;
        float b1 = (EPI >= 1) ? bias[m1] : 0.f;
        float v00 = acc00[r] + b0, v01 = acc01[r] + b0;
        float v10 = acc10[r] + b1, v11 = acc11[r] + b1;
        if (EPI == 2) {
            v00 = fmaxf(v00, 0.f); v01 = fmaxf(v01, 0.f);
            v10 = fmaxf(v10, 0.f); v11 = fmaxf(v11, 0.f);
        }
        D[(long)m0 * ldd + nb0]      = (_Float16)v00;
        D[(long)m0 * ldd + nb0 + 16] = (_Float16)v01;
        D[(long)m1 * ldd + nb0]      = (_Float16)v10;
        D[(long)m1 * ldd + nb0 + 16] = (_Float16)v11;
    }
}

// ---------------------------------------------------------------------------
// Elementwise / small kernels (f16 activations, f32 math)
// ---------------------------------------------------------------------------
__global__ void cvt_f32_f16_kernel(const float* __restrict__ s, _Float16* __restrict__ d, long n)
{
    long i = ((long)blockIdx.x * 256 + threadIdx.x) * 4;
    if (i + 4 <= n) {
        float4 f = *(const float4*)(s + i);
        d[i + 0] = (_Float16)f.x; d[i + 1] = (_Float16)f.y;
        d[i + 2] = (_Float16)f.z; d[i + 3] = (_Float16)f.w;
    } else {
        for (; i < n; ++i) d[i] = (_Float16)s[i];
    }
}

// f32 (rows x cols) -> f16 (rows x cpad), zero-filled columns cols..cpad-1
__global__ void cvt2d_pad_kernel(const float* __restrict__ s, _Float16* __restrict__ d,
                                 int rows, int cols, int cpad)
{
    long idx = (long)blockIdx.x * 256 + threadIdx.x;
    if (idx >= (long)rows * cpad) return;
    int r = (int)(idx / cpad), cc = (int)(idx % cpad);
    d[idx] = (cc < cols) ? (_Float16)s[(long)r * cols + cc] : (_Float16)0.f;
}

__global__ void rms_norm_kernel(const float* __restrict__ x, _Float16* __restrict__ y, int T)
{
    const int b = blockIdx.x;
    const float* xb = x + (long)b * T;
    _Float16* yb = y + (long)b * T;
    __shared__ float red[256];
    float s = 0.f;
    for (int i = threadIdx.x; i < T; i += 256) { float v = xb[i]; s += v * v; }
    red[threadIdx.x] = s; __syncthreads();
    for (int off = 128; off; off >>= 1) {
        if (threadIdx.x < off) red[threadIdx.x] += red[threadIdx.x + off];
        __syncthreads();
    }
    float scale = rsqrtf(red[0] / (float)T);
    for (int i = threadIdx.x; i < T; i += 256) yb[i] = (_Float16)(xb[i] * scale);
}

__global__ void diff_embed_kernel(const float* __restrict__ t,
                                  const float* __restrict__ w1, const float* __restrict__ b1,
                                  const float* __restrict__ w2, const float* __restrict__ b2,
                                  float* __restrict__ demb)
{
    const int b = blockIdx.x;
    __shared__ float h[128];
    __shared__ float h1[512];
    const int tid = threadIdx.x; // 512
    if (tid < 64) {
        float a = t[b] * powf(10.f, (float)tid * (4.f / 63.f));
        h[tid] = sinf(a);
        h[tid + 64] = cosf(a);
    }
    __syncthreads();
    {
        float s = b1[tid];
        const float* w = w1 + (long)tid * 128;
        for (int k = 0; k < 128; ++k) s += w[k] * h[k];
        h1[tid] = s / (1.f + expf(-s));
    }
    __syncthreads();
    {
        float s = b2[tid];
        const float* w = w2 + (long)tid * 512;
        for (int k = 0; k < 512; ++k) s += w[k] * h1[k];
        demb[b * 512 + tid] = s / (1.f + expf(-s));
    }
}

__global__ void dw_proj_kernel(const float* __restrict__ demb, const float* __restrict__ dw,
                               float* __restrict__ out, int Mch)
{
    int idx = blockIdx.x * 256 + threadIdx.x;
    if (idx >= 2 * Mch) return;
    int b = idx / Mch, m = idx % Mch;
    const float* d = demb + b * 512;
    const float* w = dw + (long)m * 512;
    float s = 0.f;
    for (int k = 0; k < 512; ++k) s += d[k] * w[k];
    out[idx] = s;
}

// im2col with padded row count: rows cin*K .. rowsPad-1 are written as zero.
__global__ void im2col_kernel(const _Float16* __restrict__ in, _Float16* __restrict__ out,
                              int cin, int Tin, int Tout, int K, int stride, int pad,
                              int rowsPad)
{
    long idx = (long)blockIdx.x * 256 + threadIdx.x;
    if (idx >= (long)rowsPad * Tout) return;
    int t = (int)(idx % Tout);
    int row = (int)(idx / Tout);
    _Float16 v = (_Float16)0.f;
    if (row < cin * K) {
        int kk = row % K;
        int ci = row / K;
        int ti = t * stride - pad + kk;
        if (ti >= 0 && ti < Tin) v = in[(long)ci * Tin + ti];
    }
    out[idx] = v;
}

__global__ void glu_kernel(const _Float16* __restrict__ h, const float* __restrict__ dbias,
                           _Float16* __restrict__ out, int C, long T)
{
    long idx = (long)blockIdx.x * 256 + threadIdx.x;
    if (idx >= (long)C * T) return;
    long t = idx % T;
    int  c = (int)(idx / T);
    float a = (float)h[(long)c * T + t] + dbias[c];
    float g = (float)h[(long)(c + C) * T + t] + dbias[c + C];
    out[idx] = (_Float16)(a * (1.f / (1.f + expf(-g))));
}

__global__ void add_kernel(_Float16* __restrict__ x, const _Float16* __restrict__ y, long n)
{
    long idx = (long)blockIdx.x * 256 + threadIdx.x;
    if (idx < n) x[idx] = (_Float16)((float)x[idx] + (float)y[idx]);
}

__global__ void ln_kernel(const _Float16* __restrict__ a, const _Float16* __restrict__ r,
                          const float* __restrict__ g, const float* __restrict__ be,
                          _Float16* __restrict__ out, int E, int T)
{
    const int t = blockIdx.x;
    __shared__ float s1[256], s2[256];
    float s = 0.f, q = 0.f;
    for (int e = threadIdx.x; e < E; e += 256) {
        float v = (float)a[(long)e * T + t] + (float)r[(long)e * T + t];
        s += v; q += v * v;
    }
    s1[threadIdx.x] = s; s2[threadIdx.x] = q; __syncthreads();
    for (int off = 128; off; off >>= 1) {
        if (threadIdx.x < off) {
            s1[threadIdx.x] += s1[threadIdx.x + off];
            s2[threadIdx.x] += s2[threadIdx.x + off];
        }
        __syncthreads();
    }
    float mean = s1[0] / (float)E;
    float var  = s2[0] / (float)E - mean * mean;
    float rstd = rsqrtf(var + 1e-5f);
    for (int e = threadIdx.x; e < E; e += 256) {
        float v = (float)a[(long)e * T + t] + (float)r[(long)e * T + t];
        out[(long)e * T + t] = (_Float16)((v - mean) * rstd * g[e] + be[e]);
    }
}

__global__ void qkv_bias_kernel(const _Float16* __restrict__ h, const float* __restrict__ u,
                                _Float16* __restrict__ out, int rows, int T, int hd, int three_hd)
{
    long idx = (long)blockIdx.x * 256 + threadIdx.x;
    if (idx >= (long)rows * T) return;
    int e = (int)(idx / T);
    int r = e % three_hd;
    int hh = e / three_hd;
    float bv = (r < hd) ? u[hh * hd + r] : 0.f;
    out[idx] = (_Float16)((float)h[idx] + bv);
}

// (Lp=256 rows x E); rows 2T-1..255 zero so padded GEMM columns are exact zeros.
__global__ void pe_raw_kernel(_Float16* __restrict__ pe, int T, int E, int Lp)
{
    long idx = (long)blockIdx.x * 256 + threadIdx.x;
    if (idx >= (long)Lp * E) return;
    int l = (int)(idx / E);
    int e = (int)(idx % E);
    float val = 0.f;
    if (l < 2 * T - 1) {
        int f = e >> 1;
        float pos = (float)(l - (T - 1));
        float inv = powf(10000.f, -(float)(2 * f) / (float)E);
        float ang = pos * inv;
        val = (e & 1) ? sinf(ang) : cosf(ang);
    }
    pe[idx] = (_Float16)val;
}

// rel-shift + scale + softmax over j; BD has leading dim ldbd (=256 padded).
__global__ void attn_softmax_kernel(const _Float16* __restrict__ AC, const _Float16* __restrict__ BD,
                                    _Float16* __restrict__ P, int T, int hd, int ldbd)
{
    const int i = blockIdx.x, h = blockIdx.y, j = threadIdx.x;
    const int L = 2 * T - 1, Lp = 2 * T + 1;
    __shared__ float red[128];
    long flat = (long)(2 * i + 1) * T + j;
    int ip = (int)(flat / Lp), jp = (int)(flat % Lp);
    float bd = (jp >= 1 && jp <= L) ? (float)BD[((long)h * T + ip) * ldbd + (jp - 1)] : 0.f;
    float v = ((float)AC[((long)h * T + i) * T + j] + bd) * rsqrtf((float)hd);
    red[j] = v; __syncthreads();
    for (int off = 64; off; off >>= 1) {
        if (j < off) red[j] = fmaxf(red[j], red[j + off]);
        __syncthreads();
    }
    float m = red[0]; __syncthreads();
    float e = expf(v - m);
    red[j] = e; __syncthreads();
    for (int off = 64; off; off >>= 1) {
        if (j < off) red[j] += red[j + off];
        __syncthreads();
    }
    P[((long)h * T + i) * T + j] = (_Float16)(e / red[0]);
}

// conv_transpose1d gather: stride 4, K=8, pad 2 -> exactly 2 taps per output.
__global__ void col2im_kernel(const _Float16* __restrict__ G, const float* __restrict__ bias,
                              void* __restrict__ out, int cout, int Tin, int Tout,
                              int relu, int outf32)
{
    long idx = (long)blockIdx.x * 256 + threadIdx.x;
    if (idx >= (long)cout * Tout) return;
    int t  = (int)(idx % Tout);
    int co = (int)(idx / Tout);
    float s = bias ? bias[co] : 0.f;
    int r = (t + 2) & 3;
    for (int kk = r; kk < 8; kk += 4) {
        int ti = (t + 2 - kk) >> 2;
        if (ti >= 0 && ti < Tin) s += (float)G[((long)co * 8 + kk) * Tin + ti];
    }
    if (relu) s = fmaxf(s, 0.f);
    if (outf32) ((float*)out)[idx] = s;
    else        ((_Float16*)out)[idx] = (_Float16)s;
}

// ---------------------------------------------------------------------------
// Host-side GEMM dispatch
// ---------------------------------------------------------------------------
static inline void gemm(int TA, int TB, int EPI,
                        const _Float16* A, const _Float16* B, _Float16* D, const float* bias,
                        int M, int N, int K, int lda, int ldb, int ldd,
                        long sA, long sB, long sD, int batch, hipStream_t st)
{
    dim3 grid((unsigned)(N / 64), (unsigned)(M / 64), (unsigned)batch);
    dim3 block(128);
#define GEMM_CASE(ta, tb, epi)                                                   \
    if (TA == ta && TB == tb && EPI == epi) {                                    \
        wmma_gemm_kernel<ta, tb, epi><<<grid, block, 0, st>>>(                   \
            A, B, D, bias, M, N, K, lda, ldb, ldd, sA, sB, sD);                  \
        return;                                                                  \
    }
    GEMM_CASE(0, 0, 0) GEMM_CASE(0, 0, 1) GEMM_CASE(0, 0, 2)
    GEMM_CASE(1, 0, 0) GEMM_CASE(0, 1, 0)
#undef GEMM_CASE
}

static inline unsigned gblk(long n) { return (unsigned)((n + 255) / 256); }

// ---------------------------------------------------------------------------
// Orchestration
// ---------------------------------------------------------------------------
extern "C" void kernel_launch(void* const* d_in, const int* in_sizes, int n_in,
                              void* d_out, int out_size, void* d_ws, size_t ws_size,
                              hipStream_t stream)
{
    if (n_in < 112) return;
    const float* x_in  = (const float*)d_in[0];
    const float* tstep = (const float*)d_in[1];
    auto P = [&](int i) { return (const float*)d_in[i]; };
    // Param leaf order (jax pytree, dict keys sorted):
    //   dec[i]: dw, post_b, post_w, pre_w             -> 2 + i*4 + j
    //   diff  : b1=26, b2=27, w1=28, w2=29
    //   enc[i]: b0, b1, dw, w0, w1                    -> 30 + i*5 + j
    //   layers[l]: in_w,out_w,pos_w,u,v,ff_b1,ff_b2,ff_w1,ff_w2,
    //              ln1_b,ln1_g,ln2_b,ln2_g            -> 60 + l*13 + j

    const int B = 2, Tfull = 524288, E = 2048, Ttr = 128, NH = 16, HD = 128;
    const int Lp = 256;         // padded rel-pos length (GEMM-friendly; L=255)

    // ---- workspace allocators ---------------------------------------------
    char* wsp = (char*)d_ws;
    size_t off = 0;
    auto allocB = [&](size_t bytes) -> void* {
        void* p = wsp + off;
        off += (bytes + 255) & ~(size_t)255;
        return p;
    };
    auto allocH = [&](size_t n) { return (_Float16*)allocB(n * 2); };
    auto allocF = [&](size_t n) { return (float*)allocB(n * 4); };

    float* demb = allocF((size_t)B * 512);
    float* dwp  = allocF((size_t)B * 4096);
    const size_t actMax = (size_t)B * 64 * 131072; // 16.8M elems
    _Float16* act0  = allocH(actMax);
    _Float16* act1  = allocH(actMax);
    _Float16* hbuf  = allocH(16777216);
    _Float16* hbuf2 = allocH(16777216);
    _Float16* imcol = allocH(25165824);   // also reused for conv-transpose G
    _Float16* peraw  = allocH((size_t)Lp * E);
    _Float16* peproj = allocH((size_t)E * Lp);
    _Float16* h3     = allocH((size_t)3 * E * Ttr);
    _Float16* qu     = allocH((size_t)3 * E * Ttr);
    _Float16* qv     = allocH((size_t)3 * E * Ttr);
    _Float16* ACb    = allocH((size_t)NH * Ttr * Ttr);
    _Float16* BDb    = allocH((size_t)NH * Ttr * Lp);
    _Float16* Pb     = allocH((size_t)NH * Ttr * Ttr);
    _Float16* ctx    = allocH((size_t)E * Ttr);
    _Float16* ffh    = allocH((size_t)4 * E * Ttr);
    _Float16* tmpE   = allocH((size_t)E * Ttr);
    _Float16* saved[6];
    for (int i = 0; i < 6; ++i) {
        int c = 64 << i, Tout = Tfull >> (2 * i + 2);
        saved[i] = allocH((size_t)B * c * Tout);
    }
    // reusable per-layer f16 weight scratch (max: transformer layer, 13*E^2 f16)
    char* wscr = (char*)allocB((size_t)128 << 20);
    size_t woff = 0;
    auto wreset = [&]() { woff = 0; };
    auto wcvt = [&](const float* src, size_t n) -> const _Float16* {
        _Float16* d = (_Float16*)(wscr + woff);
        woff += (n * 2 + 255) & ~(size_t)255;
        cvt_f32_f16_kernel<<<gblk((long)((n + 3) / 4)), 256, 0, stream>>>(src, d, (long)n);
        return d;
    };
    auto wcvt2d = [&](const float* src, int rows, int cols, int cpad) -> const _Float16* {
        _Float16* d = (_Float16*)(wscr + woff);
        woff += ((size_t)rows * cpad * 2 + 255) & ~(size_t)255;
        cvt2d_pad_kernel<<<gblk((long)rows * cpad), 256, 0, stream>>>(src, d, rows, cols, cpad);
        return d;
    };

    // ---- diffusion embedding + input RMS norm -----------------------------
    diff_embed_kernel<<<B, 512, 0, stream>>>(tstep, P(28), P(26), P(29), P(27), demb);
    rms_norm_kernel<<<B, 256, 0, stream>>>(x_in, act0, Tfull);

    // ---- encoder ----------------------------------------------------------
    const _Float16* xcur = act0;
    for (int i = 0; i < 6; ++i) {
        const int cin  = (i == 0) ? 1 : (64 << (i - 1));
        const int c    = 64 << i;
        const int Tin  = Tfull >> (2 * i);
        const int Tout = Tin / 4;
        const int Kc   = cin * 8;                  // conv0 contraction size
        const int Kp   = (Kc < 32) ? 32 : Kc;      // padded to >=32 (layer 0)
        const float* b0 = P(30 + i * 5 + 0);
        const float* b1 = P(30 + i * 5 + 1);
        const float* dw = P(30 + i * 5 + 2);
        wreset();
        const _Float16* w0h = (Kp == Kc)
            ? wcvt(P(30 + i * 5 + 3), (size_t)c * Kc)
            : wcvt2d(P(30 + i * 5 + 3), c, Kc, Kp);
        const _Float16* w1h = wcvt(P(30 + i * 5 + 4), (size_t)2 * c * c);

        dw_proj_kernel<<<gblk(2L * 2 * c), 256, 0, stream>>>(demb, dw, dwp, 2 * c);

        for (int b = 0; b < B; ++b) {
            const _Float16* xb = xcur + (size_t)b * cin * Tin;
            im2col_kernel<<<gblk((long)Kp * Tout), 256, 0, stream>>>(
                xb, imcol, cin, Tin, Tout, 8, 4, 2, Kp);
            gemm(0, 0, 2, w0h, imcol, hbuf, b0, c, Tout, Kp,
                 Kp, Tout, Tout, 0, 0, 0, 1, stream);
            gemm(0, 0, 1, w1h, hbuf, hbuf2, b1, 2 * c, Tout, c,
                 c, Tout, Tout, 0, 0, 0, 1, stream);
            glu_kernel<<<gblk((long)c * Tout), 256, 0, stream>>>(
                hbuf2, dwp + (size_t)b * 2 * c,
                saved[i] + (size_t)b * c * Tout, c, Tout);
        }
        xcur = saved[i];
    }

    // ---- transformer ------------------------------------------------------
    hipMemcpyAsync(act0, saved[5], (size_t)B * E * Ttr * sizeof(_Float16),
                   hipMemcpyDeviceToDevice, stream);
    pe_raw_kernel<<<gblk((long)Lp * E), 256, 0, stream>>>(peraw, Ttr, E, Lp);

    for (int l = 0; l < 4; ++l) {
        const int base = 60 + l * 13;
        wreset();
        const _Float16* in_wh  = wcvt(P(base + 0), (size_t)3 * E * E);
        const _Float16* out_wh = wcvt(P(base + 1), (size_t)E * E);
        const _Float16* pos_wh = wcvt(P(base + 2), (size_t)E * E);
        const _Float16* ffw1h  = wcvt(P(base + 7), (size_t)4 * E * E);
        const _Float16* ffw2h  = wcvt(P(base + 8), (size_t)4 * E * E);
        const float* ub   = P(base + 3);
        const float* vb   = P(base + 4);
        const float* ffb1 = P(base + 5);
        const float* ffb2 = P(base + 6);
        const float* ln1b = P(base + 9);
        const float* ln1g = P(base + 10);
        const float* ln2b = P(base + 11);
        const float* ln2g = P(base + 12);

        // pe_proj (E x Lp) = pos_w @ peraw^T  (peraw is (Lp,E) -> TB=1)
        gemm(0, 1, 0, pos_wh, peraw, peproj, nullptr, E, Lp, E,
             E, E, Lp, 0, 0, 0, 1, stream);

        for (int b = 0; b < B; ++b) {
            _Float16* xb = act0 + (size_t)b * E * Ttr;
            gemm(0, 0, 0, in_wh, xb, h3, nullptr, 3 * E, Ttr, E,
                 E, Ttr, Ttr, 0, 0, 0, 1, stream);
            qkv_bias_kernel<<<gblk((long)3 * E * Ttr), 256, 0, stream>>>(
                h3, ub, qu, 3 * E, Ttr, HD, 3 * HD);
            qkv_bias_kernel<<<gblk((long)3 * E * Ttr), 256, 0, stream>>>(
                h3, vb, qv, 3 * E, Ttr, HD, 3 * HD);
            // AC[h] = (q+u)^T k, batched over heads (q/k/v interleaved per head)
            gemm(1, 0, 0, qu, h3 + (size_t)HD * Ttr, ACb, nullptr,
                 Ttr, Ttr, HD, Ttr, Ttr, Ttr,
                 (long)3 * HD * Ttr, (long)3 * HD * Ttr, (long)Ttr * Ttr, NH, stream);
            // BD[h] = (q+v)^T pe_h   (padded column Lp-1 is exactly zero)
            gemm(1, 0, 0, qv, peproj, BDb, nullptr,
                 Ttr, Lp, HD, Ttr, Lp, Lp,
                 (long)3 * HD * Ttr, (long)HD * Lp, (long)Ttr * Lp, NH, stream);
            attn_softmax_kernel<<<dim3(Ttr, NH), Ttr, 0, stream>>>(
                ACb, BDb, Pb, Ttr, HD, Lp);
            // ctx[h] = v_h @ P^T
            gemm(0, 1, 0, h3 + (size_t)2 * HD * Ttr, Pb, ctx, nullptr,
                 HD, Ttr, Ttr, Ttr, Ttr, Ttr,
                 (long)3 * HD * Ttr, (long)Ttr * Ttr, (long)HD * Ttr, NH, stream);
            gemm(0, 0, 0, out_wh, ctx, tmpE, nullptr, E, Ttr, E,
                 E, Ttr, Ttr, 0, 0, 0, 1, stream);
            ln_kernel<<<Ttr, 256, 0, stream>>>(tmpE, xb, ln1g, ln1b, xb, E, Ttr);
            gemm(0, 0, 2, ffw1h, xb, ffh, ffb1, 4 * E, Ttr, E,
                 E, Ttr, Ttr, 0, 0, 0, 1, stream);
            gemm(0, 0, 1, ffw2h, ffh, tmpE, ffb2, E, Ttr, 4 * E,
                 4 * E, Ttr, Ttr, 0, 0, 0, 1, stream);
            ln_kernel<<<Ttr, 256, 0, stream>>>(tmpE, xb, ln2g, ln2b, xb, E, Ttr);
        }
    }

    // ---- decoder ----------------------------------------------------------
    _Float16* decIn  = act0;
    _Float16* decOut = act1;
    for (int i = 0; i < 6; ++i) {
        const int c    = 2048 >> i;
        const int cout = (i == 5) ? 1 : c / 2;
        const int Tin  = 128 << (2 * i);
        const int Tout = Tin * 4;
        const int Mg   = cout * 8;                 // G rows
        const int Mgp  = (Mg < 64) ? 64 : Mg;      // padded to >=64 (layer 5)
        const float* dw     = P(2 + i * 4 + 0);
        const float* post_b = P(2 + i * 4 + 1);
        wreset();
        const _Float16* post_wh = (Mgp == Mg)
            ? wcvt(P(2 + i * 4 + 2), (size_t)c * Mg)
            : wcvt2d(P(2 + i * 4 + 2), c, Mg, Mgp);
        const _Float16* pre_wh  = wcvt(P(2 + i * 4 + 3), (size_t)2 * c * c * 3);

        add_kernel<<<gblk((long)B * c * Tin), 256, 0, stream>>>(
            decIn, saved[5 - i], (long)B * c * Tin);
        dw_proj_kernel<<<gblk(2L * 2 * c), 256, 0, stream>>>(demb, dw, dwp, 2 * c);

        for (int b = 0; b < B; ++b) {
            const _Float16* xb = decIn + (size_t)b * c * Tin;
            im2col_kernel<<<gblk((long)c * 3 * Tin), 256, 0, stream>>>(
                xb, imcol, c, Tin, Tin, 3, 1, 1, c * 3);
            gemm(0, 0, 0, pre_wh, imcol, hbuf, nullptr, 2 * c, Tin, 3 * c,
                 3 * c, Tin, Tin, 0, 0, 0, 1, stream);
            glu_kernel<<<gblk((long)c * Tin), 256, 0, stream>>>(
                hbuf, dwp + (size_t)b * 2 * c, hbuf2, c, Tin);
            // G (Mgp x T) = post_w^T-view @ g  (post_w is (c, Mg)->padded, TA=1)
            gemm(1, 0, 0, post_wh, hbuf2, imcol, nullptr, Mgp, Tin, c,
                 Mgp, Tin, Tin, 0, 0, 0, 1, stream);
            void* ob = (i == 5) ? (void*)((float*)d_out + (size_t)b * Tout)
                                : (void*)(decOut + (size_t)b * cout * Tout);
            col2im_kernel<<<gblk((long)cout * Tout), 256, 0, stream>>>(
                imcol, post_b, ob, cout, Tin, Tout, (i < 5) ? 1 : 0, (i == 5) ? 1 : 0);
        }
        _Float16* t = decIn; decIn = decOut; decOut = t;
    }
    (void)in_sizes; (void)out_size; (void)ws_size;
}
